// LSTM_84937273246322
// MI455X (gfx1250) — matrix operand order repaired and checked by reading, not compiled, not run on hardware
//
#include <hip/hip_runtime.h>
#include <hip/hip_bf16.h>

typedef __bf16 bf16_t;
typedef __attribute__((ext_vector_type(16))) __bf16 v16bf;
typedef __attribute__((ext_vector_type(8)))  __bf16 v8bf;
typedef __attribute__((ext_vector_type(8)))  float  v8f;
typedef __attribute__((ext_vector_type(4)))  unsigned int v4u;
typedef __attribute__((ext_vector_type(8)))  int v8i;
typedef __attribute__((ext_vector_type(4)))  int v4i;

#define TT 512
#define BB 64
#define EE 512
#define HH 1024

// -------- workspace layout (bytes), all offsets multiples of 256 --------
static const size_t CNT_OFF = 0;                                   // barrier counter (256 B reserved)
static const size_t S1_OFF  = 256;                                 // state buf 1 (read at t=0) [B,H] bf16
static const size_t S0_OFF  = S1_OFF + (size_t)BB * HH * 2;        // state buf 0
static const size_t EMB_OFF = S0_OFF + (size_t)BB * HH * 2;        // embeds bf16 [T,B,E]
static const size_t WGX_OFF = EMB_OFF + (size_t)TT * BB * EE * 2;
static const size_t WIX_OFF = WGX_OFF + (size_t)HH * EE * 2;
static const size_t WOX_OFF = WIX_OFF + (size_t)HH * EE * 2;
static const size_t WGH_OFF = WOX_OFF + (size_t)HH * EE * 2;
static const size_t WIH_OFF = WGH_OFF + (size_t)HH * HH * 2;
static const size_t WOH_OFF = WIH_OFF + (size_t)HH * HH * 2;

// -------- fp32 -> bf16 conversion (grid-stride) --------
__global__ void cvt_f32_to_bf16(const float* __restrict__ src, bf16_t* __restrict__ dst, int n) {
    int i = blockIdx.x * blockDim.x + threadIdx.x;
    int stride = gridDim.x * blockDim.x;
    for (; i < n; i += stride) dst[i] = (bf16_t)src[i];
}

__device__ __forceinline__ float sigm_f(float x) {
    return 1.0f / (1.0f + __expf(-x));
}
__device__ __forceinline__ float tanh_f(float x) {
    x = fminf(fmaxf(x, -15.0f), 15.0f);
    float e = __expf(2.0f * x);
    return (e - 1.0f) / (e + 1.0f);
}

// A-fragment: lane L holds row M=L&15; VGPRs 0-3 hold K=(L>>4)*8+0..7,
// VGPRs 4-7 hold K=16+(L>>4)*8+0..7 -> two contiguous 8-bf16 chunks 16 apart.
__device__ __forceinline__ v16bf load_a(const bf16_t* p) {
    v8bf lo = *(const v8bf*)(p);
    v8bf hi = *(const v8bf*)(p + 16);
    return __builtin_shufflevector(lo, hi, 0,1,2,3,4,5,6,7,8,9,10,11,12,13,14,15);
}

#define WMMA_BF16(acc, a, b) \
    __builtin_amdgcn_wmma_f32_16x16x32_bf16(false, (a), false, (b), (short)0, (acc), false, false)

// ---- Tensor Data Mover: 2D tile (nrows x rowlen bf16 elems) global -> LDS ----
// D# per CDNA5 ISA: group0 = {ctrl, lds_addr, gaddr lo/hi+type},
// group1 = {mask/data_size, dims/tiles/strides}; groups 2,3 zero => 2D tensor.
// amdgpu-toolchain (clang-23) 6-arg builtin: (v4u, v8i, v4i, v4i, v8i, i32 cpol).
__device__ __forceinline__ void tdm_load_2d(unsigned lds_off, const void* gptr,
                                            unsigned rowlen, unsigned nrows,
                                            unsigned row_stride) {
    unsigned long long ga = (unsigned long long)(size_t)gptr;
    v4u g0;
    g0[0] = 1u;                                               // count=1 (valid), user mode
    g0[1] = lds_off;                                          // lds_addr (bytes)
    g0[2] = (unsigned)(ga & 0xffffffffu);                     // global_addr[31:0]
    g0[3] = (unsigned)((ga >> 32) & 0x01ffffffu) | (2u << 30);// global_addr[56:32], type=2
    v8i g1;
    g1[0] = (int)(1u << 16);                                  // wg_mask=0, data_size=1 (2B)
    g1[1] = (int)((rowlen & 0xffffu) << 16);                  // tensor_dim0[15:0]
    g1[2] = (int)(((rowlen >> 16) & 0xffffu) | ((nrows & 0xffffu) << 16)); // dim0 hi | dim1 lo
    g1[3] = (int)(((nrows >> 16) & 0xffffu) | ((rowlen & 0xffffu) << 16)); // dim1 hi | tile_dim0
    g1[4] = (int)(nrows & 0xffffu);                           // tile_dim1 | tile_dim2=0
    g1[5] = (int)row_stride;                                  // tensor_dim0_stride[31:0]
    g1[6] = 0;                                                // stride hi | dim1_stride lo
    g1[7] = 0;                                                // dim1_stride hi
    v4i z4 = {0, 0, 0, 0};
    v8i z8 = {0, 0, 0, 0, 0, 0, 0, 0};
    __builtin_amdgcn_tensor_load_to_lds(g0, g1, z4, z4, z8, 0);
}

#define LDS_X_ELEMS (16 * EE)            // x_t tile: 16 cols x 512
#define LDS_TOTAL   (16 * EE + 16 * HH)  // + s tile: 16 cols x 1024  (48 KB)

// -------- persistent fused LSTM recurrence --------
// grid = 32 blocks x 256 threads = 256 waves; wave owns one 16x16 (h x b) tile,
// computes gates g/i/o fused over K = E (input proj) + H (recurrent).
// B operands (shared by the 8 waves of a block) staged via TDM into LDS.
__global__ __launch_bounds__(256, 1) void lstm_persistent(
    const bf16_t* __restrict__ embb,
    const bf16_t* __restrict__ Wgx, const bf16_t* __restrict__ Wgh, const float* __restrict__ bg,
    const bf16_t* __restrict__ Wix, const bf16_t* __restrict__ Wih, const float* __restrict__ bi,
    const bf16_t* __restrict__ Wox, const bf16_t* __restrict__ Woh, const float* __restrict__ bo,
    bf16_t* __restrict__ s0, bf16_t* __restrict__ s1,
    float* __restrict__ out, unsigned* __restrict__ cnt)
{
    __shared__ __align__(64) bf16_t lds_buf[LDS_TOTAL];

    const int lane = threadIdx.x & 31;
    const int wave = threadIdx.x >> 5;
    const int b0   = (blockIdx.x & 3) * 16;                    // 4 b-tiles
    const int h0   = ((blockIdx.x >> 2) * 8 + wave) * 16;      // 64 h-tiles
    const int n    = lane & 15;        // B-fragment column / D column
    const int half = lane >> 4;
    const int kA   = half * 8;         // A-fragment K sub-offset
    const int kB   = half * 8;         // B-fragment K sub-offset
    const int mB   = h0 + half * 8;    // first of 8 contiguous D rows for this lane

    // Row pointers for A operands (weights are row-major [H, K])
    const bf16_t* Agx = Wgx + (size_t)(h0 + n) * EE + kA;
    const bf16_t* Aix = Wix + (size_t)(h0 + n) * EE + kA;
    const bf16_t* Aox = Wox + (size_t)(h0 + n) * EE + kA;
    const bf16_t* Agh = Wgh + (size_t)(h0 + n) * HH + kA;
    const bf16_t* Aih = Wih + (size_t)(h0 + n) * HH + kA;
    const bf16_t* Aoh = Woh + (size_t)(h0 + n) * HH + kA;

    // LDS B-fragment pointers (column n, K sub-offset kB)
    const bf16_t* BxL = lds_buf + (size_t)n * EE + kB;
    const bf16_t* BsL = lds_buf + LDS_X_ELEMS + (size_t)n * HH + kB;
    const unsigned lds_base = (unsigned)(size_t)(void*)lds_buf; // LDS byte offset (addr[31:0])

    // Bias pre-loaded into accumulator init (D rows are contiguous per lane)
    const v8f biasg = *(const v8f*)(bg + mB);
    const v8f biasi = *(const v8f*)(bi + mB);
    const v8f biaso = *(const v8f*)(bo + mB);

    const unsigned nb = gridDim.x;

    for (int t = 0; t < TT; ++t) {
        const bf16_t* sR = (t & 1) ? s0 : s1;   // read s_{t-1} (transposed [B,H] bf16)
        bf16_t*       sW = (t & 1) ? s1 : s0;   // write s_t

        // ---- stage B operands into LDS with the Tensor Data Mover ----
        if (wave == 0) {
            tdm_load_2d(lds_base,
                        embb + ((size_t)t * BB + b0) * EE, EE, 16, EE);
            tdm_load_2d(lds_base + LDS_X_ELEMS * 2,
                        sR + (size_t)b0 * HH, HH, 16, HH);
            __builtin_amdgcn_s_wait_tensorcnt(0);
        }
        __syncthreads();

        v8f ag = biasg, ai = biasi, ao = biaso;

        // ---- input projection part: K over E ----
        #pragma unroll 4
        for (int k = 0; k < EE; k += 32) {
            v16bf bv = load_a(BxL + k);
            ag = WMMA_BF16(ag, load_a(Agx + k), bv);
            ai = WMMA_BF16(ai, load_a(Aix + k), bv);
            ao = WMMA_BF16(ao, load_a(Aox + k), bv);
        }
        // ---- recurrent part: K over H ----
        #pragma unroll 4
        for (int k = 0; k < HH; k += 32) {
            v16bf bv = load_a(BsL + k);
            ag = WMMA_BF16(ag, load_a(Agh + k), bv);
            ai = WMMA_BF16(ai, load_a(Aih + k), bv);
            ao = WMMA_BF16(ao, load_a(Aoh + k), bv);
        }

        // ---- activations: c = tanh(g)*sigm(i); h = tanh(c)*sigm(o) ----
        v8f  hv;
        v8bf cv;
        #pragma unroll
        for (int r = 0; r < 8; ++r) {
            float g  = tanh_f(ag[r]);
            float ig = sigm_f(ai[r]);
            float og = sigm_f(ao[r]);
            float c  = g * ig;
            cv[r] = (bf16_t)c;
            hv[r] = tanh_f(c) * og;
        }

        // state store (transposed, bf16): 8 contiguous elems per lane
        *(v8bf*)(sW + (size_t)(b0 + n) * HH + mB) = cv;
        // output store out[t, b, h] fp32: 8 contiguous floats per lane
        *(v8f*)(out + ((size_t)t * BB + b0 + n) * HH + mB) = hv;

        // ---- device-wide barrier between timesteps ----
        __threadfence();
        __syncthreads();
        if (threadIdx.x == 0) {
            __hip_atomic_fetch_add(cnt, 1u, __ATOMIC_RELEASE, __HIP_MEMORY_SCOPE_AGENT);
            const unsigned target = nb * (unsigned)(t + 1);
            while (__hip_atomic_load(cnt, __ATOMIC_ACQUIRE, __HIP_MEMORY_SCOPE_AGENT) < target)
                __builtin_amdgcn_s_sleep(2);
        }
        __syncthreads();
        __threadfence();
    }
}

extern "C" void kernel_launch(void* const* d_in, const int* in_sizes, int n_in,
                              void* d_out, int out_size, void* d_ws, size_t ws_size,
                              hipStream_t stream) {
    (void)in_sizes; (void)n_in; (void)out_size; (void)ws_size;
    const float* embeds = (const float*)d_in[0];
    const float* Wgx = (const float*)d_in[1];
    const float* Wgh = (const float*)d_in[2];
    const float* bg  = (const float*)d_in[3];
    const float* Wix = (const float*)d_in[4];
    const float* Wih = (const float*)d_in[5];
    const float* bi  = (const float*)d_in[6];
    // f gate (indices 7,8,9) is dead: multiplies prev_c == 0 everywhere.
    const float* Wox = (const float*)d_in[10];
    const float* Woh = (const float*)d_in[11];
    const float* bo  = (const float*)d_in[12];
    float* out = (float*)d_out;

    char* ws = (char*)d_ws;
    unsigned* cnt = (unsigned*)(ws + CNT_OFF);
    bf16_t* s1  = (bf16_t*)(ws + S1_OFF);
    bf16_t* s0  = (bf16_t*)(ws + S0_OFF);
    bf16_t* emb = (bf16_t*)(ws + EMB_OFF);
    bf16_t* wgx = (bf16_t*)(ws + WGX_OFF);
    bf16_t* wix = (bf16_t*)(ws + WIX_OFF);
    bf16_t* wox = (bf16_t*)(ws + WOX_OFF);
    bf16_t* wgh = (bf16_t*)(ws + WGH_OFF);
    bf16_t* wih = (bf16_t*)(ws + WIH_OFF);
    bf16_t* woh = (bf16_t*)(ws + WOH_OFF);

    // Zero barrier counter + t=0 read-state buffer (deterministic each launch).
    (void)hipMemsetAsync(d_ws, 0, S0_OFF, stream);

    // fp32 -> bf16 conversions
    cvt_f32_to_bf16<<<1024, 256, 0, stream>>>(embeds, emb, TT * BB * EE);
    cvt_f32_to_bf16<<<512,  256, 0, stream>>>(Wgx, wgx, HH * EE);
    cvt_f32_to_bf16<<<512,  256, 0, stream>>>(Wix, wix, HH * EE);
    cvt_f32_to_bf16<<<512,  256, 0, stream>>>(Wox, wox, HH * EE);
    cvt_f32_to_bf16<<<512,  256, 0, stream>>>(Wgh, wgh, HH * HH);
    cvt_f32_to_bf16<<<512,  256, 0, stream>>>(Wih, wih, HH * HH);
    cvt_f32_to_bf16<<<512,  256, 0, stream>>>(Woh, woh, HH * HH);

    // Persistent recurrence: 32 blocks x 256 threads (256 waves, co-resident)
    lstm_persistent<<<32, 256, 0, stream>>>(
        emb, wgx, wgh, bg, wix, wih, bi, wox, woh, bo,
        s0, s1, out, cnt);
}
